// Mamba_38646115730132
// MI455X (gfx1250) — matrix operand order repaired and checked by reading, not compiled, not run on hardware
//
#include <hip/hip_runtime.h>

#define BATCH   2
#define SEQ     1024
#define DMODEL  1024
#define NLAYER  4
#define VOCAB   32000
#define DSTATE  16
#define DCONV   4
#define DINNER  2048
#define DTRANK  64
#define ROWS    (BATCH * SEQ)   // 2048 token rows
#define XP_PAD  128             // x_proj N padded 96 -> 128 so all GEMMs have N%64==0

typedef __bf16 bf16_t;
typedef __attribute__((ext_vector_type(16))) __bf16 v16bf;
typedef __attribute__((ext_vector_type(8)))  __bf16 v8bf;
typedef __attribute__((ext_vector_type(8)))  float  v8f;

// ---------------------------------------------------------------- helpers
__device__ __forceinline__ bf16_t f2bf(float f) {
  unsigned u = __builtin_bit_cast(unsigned, f);
  unsigned r = u + 0x7fffu + ((u >> 16) & 1u);     // round-to-nearest-even
  unsigned short h = (unsigned short)(r >> 16);
  return __builtin_bit_cast(bf16_t, h);
}
__device__ __forceinline__ float siluf(float x) { return x / (1.0f + __expf(-x)); }
__device__ __forceinline__ float softplusf(float x) {
  return (x > 20.0f) ? x : log1pf(__expf(x));
}

// ---------------------------------------------------------------- converters
__global__ void f32_to_bf16_kernel(const float* __restrict__ src,
                                   bf16_t* __restrict__ dst, long n) {
  long i = (long)blockIdx.x * blockDim.x + threadIdx.x;
  if (i < n) dst[i] = f2bf(src[i]);
}

// convert [nrows_src, K] and zero-pad rows up to nrows_dst
__global__ void f32_to_bf16_pad_kernel(const float* __restrict__ src,
                                       bf16_t* __restrict__ dst,
                                       int nrows_src, int K, long ntotal) {
  long i = (long)blockIdx.x * blockDim.x + threadIdx.x;
  if (i >= ntotal) return;
  long r = i / K;
  dst[i] = (r < nrows_src) ? f2bf(src[i]) : __builtin_bit_cast(bf16_t, (unsigned short)0);
}

// take `cols` leading columns out of rows with stride src_stride
__global__ void f32_to_bf16_strided_kernel(const float* __restrict__ src,
                                           bf16_t* __restrict__ dst,
                                           int rows, int cols, int src_stride) {
  int i = blockIdx.x * blockDim.x + threadIdx.x;
  if (i >= rows * cols) return;
  int r = i / cols, c = i % cols;
  dst[i] = f2bf(src[(long)r * src_stride + c]);
}

// ---------------------------------------------------------------- embedding
__global__ void embed_gather_kernel(const int* __restrict__ tokens,
                                    const float* __restrict__ embed,
                                    float* __restrict__ H) {
  int row = blockIdx.x;                 // 0..ROWS-1
  int tok = tokens[row];
  const float* src = embed + (long)tok * DMODEL;
  float* dst = H + (long)row * DMODEL;
  for (int i = threadIdx.x; i < DMODEL; i += blockDim.x) dst[i] = src[i];
}

// ---------------------------------------------------------------- rmsnorm -> bf16
__global__ void rmsnorm_bf16_kernel(const float* __restrict__ X,
                                    const float* __restrict__ w,
                                    bf16_t* __restrict__ out, int D) {
  int row = blockIdx.x;
  const float* x = X + (long)row * D;
  __shared__ float red[8];
  float s = 0.0f;
  for (int i = threadIdx.x; i < D; i += blockDim.x) { float v = x[i]; s += v * v; }
  for (int off = 16; off > 0; off >>= 1) s += __shfl_down(s, off, 32);
  if ((threadIdx.x & 31) == 0) red[threadIdx.x >> 5] = s;
  __syncthreads();
  if (threadIdx.x == 0) {
    float t = 0.0f;
    for (int i = 0; i < (int)(blockDim.x >> 5); ++i) t += red[i];
    red[0] = t;
  }
  __syncthreads();
  float scale = rsqrtf(red[0] / (float)D + 1e-5f);
  bf16_t* o = out + (long)row * D;
  for (int i = threadIdx.x; i < D; i += blockDim.x) o[i] = f2bf(x[i] * scale * w[i]);
}

// ---------------------------------------------------------------- bf16 WMMA GEMM
// C[M,N] (f32) = A[M,K](bf16,row-major) * W[N,K](bf16,row-major)^T   (+C if ADD)
// REQUIRES: M%16==0, N%64==0, K%32==0.
// one wave -> 16x64 tile, double-buffered fragments; 8 waves/block;
// grid = (N/64, ceil(M/128))
#define WMMA_BF16(a, b, c) __builtin_amdgcn_wmma_f32_16x16x32_bf16( \
    false, (a), false, (b), (short)0, (c), false, false)

template <bool ADD>
__global__ void gemm_bf16_wmma_kernel(const bf16_t* __restrict__ A,
                                      const bf16_t* __restrict__ W,
                                      float* __restrict__ C,
                                      int M, int N, int K) {
  const int lane  = threadIdx.x & 31;
  const int wave  = threadIdx.x >> 5;
  const int mbase = (blockIdx.y * 8 + wave) * 16;
  const int nbase = blockIdx.x * 64;
  if (mbase >= M) return;

  const int half = lane >> 4;      // 0: lanes 0-15, 1: lanes 16-31
  const int l16  = lane & 15;
  // A fragment: row = mbase+l16; K run [kk + half*16, +16)
  const bf16_t* Ap = A + (long)(mbase + l16) * K + half * 16;
  // B fragment t: column = nbase + t*16 + l16; K run [kk + half*16, +16)
  const bf16_t* Wp = W + (long)(nbase + l16) * K + half * 16;
  const long ws = (long)16 * K;    // stride between 16-column subtiles

  v8f acc0 = (v8f)(0.0f), acc1 = (v8f)(0.0f), acc2 = (v8f)(0.0f), acc3 = (v8f)(0.0f);

  auto loadA = [&](int kk) -> v16bf {
    v8bf a0 = *(const v8bf*)(Ap + kk);
    v8bf a1 = *(const v8bf*)(Ap + kk + 8);
    v16bf f;
    for (int i = 0; i < 8; ++i) { f[i] = a0[i]; f[i + 8] = a1[i]; }
    return f;
  };

  // prologue: fragments for kk = 0
  v16bf a_c = loadA(0);
  v16bf b0 = *(const v16bf*)(Wp);
  v16bf b1 = *(const v16bf*)(Wp + ws);
  v16bf b2 = *(const v16bf*)(Wp + 2 * ws);
  v16bf b3 = *(const v16bf*)(Wp + 3 * ws);

  for (int kk = 32; kk < K; kk += 32) {
    // issue next iteration's loads before consuming current registers
    v16bf a_n = loadA(kk);
    v16bf n0 = *(const v16bf*)(Wp + kk);
    v16bf n1 = *(const v16bf*)(Wp + ws + kk);
    v16bf n2 = *(const v16bf*)(Wp + 2 * ws + kk);
    v16bf n3 = *(const v16bf*)(Wp + 3 * ws + kk);
    acc0 = WMMA_BF16(a_c, b0, acc0);
    acc1 = WMMA_BF16(a_c, b1, acc1);
    acc2 = WMMA_BF16(a_c, b2, acc2);
    acc3 = WMMA_BF16(a_c, b3, acc3);
    a_c = a_n; b0 = n0; b1 = n1; b2 = n2; b3 = n3;
  }
  acc0 = WMMA_BF16(a_c, b0, acc0);
  acc1 = WMMA_BF16(a_c, b1, acc1);
  acc2 = WMMA_BF16(a_c, b2, acc2);
  acc3 = WMMA_BF16(a_c, b3, acc3);

  // store: VGPR r of each acc holds row mbase + half*8 + r, col nbase + t*16 + l16
  const int mrow0 = mbase + half * 8;
  float* Crow = C + (long)mrow0 * N + nbase + l16;
  for (int r = 0; r < 8; ++r) {
    float* p = Crow + (long)r * N;
    if (ADD) {
      p[0] += acc0[r]; p[16] += acc1[r]; p[32] += acc2[r]; p[48] += acc3[r];
    } else {
      p[0]  = acc0[r]; p[16]  = acc1[r]; p[32]  = acc2[r]; p[48]  = acc3[r];
    }
  }
}

// ---------------------------------------------------------------- causal depthwise conv + SiLU
__global__ void conv_silu_kernel(const float* __restrict__ XZ,
                                 const float* __restrict__ cw,   // [DINNER,4]
                                 const float* __restrict__ cb,   // [DINNER]
                                 float* __restrict__ XC,
                                 bf16_t* __restrict__ XCBF) {
  int idx = blockIdx.x * blockDim.x + threadIdx.x;    // over ROWS*DINNER
  if (idx >= ROWS * DINNER) return;
  int c  = idx & (DINNER - 1);
  int bl = idx >> 11;                                  // DINNER = 2048 = 2^11
  int l  = bl & (SEQ - 1);
  float acc = cb[c];
  for (int j = 0; j < DCONV; ++j) {
    int li = l + j - (DCONV - 1);
    if (li >= 0) acc += XZ[(long)(bl + j - (DCONV - 1)) * (2 * DINNER) + c] * cw[c * DCONV + j];
  }
  float s = siluf(acc);
  XC[idx] = s;
  XCBF[idx] = f2bf(s);
}

// ---------------------------------------------------------------- selective-state scan
// one thread per (batch, channel); h[16] in registers; B_t/C_t via LDS broadcast
__global__ void ssm_scan_kernel(const float* __restrict__ XC,     // [ROWS,DINNER]
                                const float* __restrict__ XDBL,   // [ROWS,XP_PAD]
                                const float* __restrict__ DTRAW,  // [ROWS,DINNER]
                                const float* __restrict__ dt_b,   // [DINNER]
                                const float* __restrict__ A_log,  // [DINNER,16]
                                const float* __restrict__ D_skip, // [DINNER]
                                const float* __restrict__ XZ,     // [ROWS,2*DINNER]
                                bf16_t* __restrict__ YBF) {       // [ROWS,DINNER]
  const int c = blockIdx.x * blockDim.x + threadIdx.x;  // channel
  const int b = blockIdx.y;                             // batch
  float A[DSTATE];
  for (int s = 0; s < DSTATE; ++s) A[s] = -__expf(A_log[c * DSTATE + s]);
  const float dtb = dt_b[c];
  const float dsk = D_skip[c];
  float h[DSTATE];
  for (int s = 0; s < DSTATE; ++s) h[s] = 0.0f;

  __shared__ float sB[DSTATE], sC[DSTATE];
  for (int l = 0; l < SEQ; ++l) {
    const long bl = (long)b * SEQ + l;
    __syncthreads();
    if (threadIdx.x < 32) {
      float v = XDBL[bl * XP_PAD + DTRANK + threadIdx.x];
      if (threadIdx.x < DSTATE) sB[threadIdx.x] = v;
      else                      sC[threadIdx.x - DSTATE] = v;
    }
    __syncthreads();
    const float dt  = softplusf(DTRAW[bl * DINNER + c] + dtb);
    const float xt  = XC[bl * DINNER + c];
    const float dtx = dt * xt;
    float y = 0.0f;
    for (int s = 0; s < DSTATE; ++s) {
      float dA = __expf(dt * A[s]);
      h[s] = dA * h[s] + dtx * sB[s];
      y += h[s] * sC[s];
    }
    y += xt * dsk;
    const float z = XZ[bl * (2 * DINNER) + DINNER + c];
    y *= siluf(z);
    YBF[bl * DINNER + c] = f2bf(y);
  }
}

// ---------------------------------------------------------------- launch
static inline void launch_gemm(const bf16_t* A, const bf16_t* W, float* C,
                               int M, int N, int K, bool add, hipStream_t s) {
  dim3 grid(N / 64, (M + 127) / 128);
  if (add) gemm_bf16_wmma_kernel<true ><<<grid, 256, 0, s>>>(A, W, C, M, N, K);
  else     gemm_bf16_wmma_kernel<false><<<grid, 256, 0, s>>>(A, W, C, M, N, K);
}
static inline void launch_cvt(const float* src, bf16_t* dst, long n, hipStream_t s) {
  f32_to_bf16_kernel<<<dim3((unsigned)((n + 255) / 256)), 256, 0, s>>>(src, dst, n);
}

extern "C" void kernel_launch(void* const* d_in, const int* in_sizes, int n_in,
                              void* d_out, int out_size, void* d_ws, size_t ws_size,
                              hipStream_t stream) {
  (void)in_sizes; (void)n_in; (void)out_size; (void)ws_size;
  const int*   tokens    = (const int*)  d_in[0];
  const float* embed     = (const float*)d_in[1];
  const float* norm_w    = (const float*)d_in[2];
  const float* in_proj_w = (const float*)d_in[3];
  const float* conv_w    = (const float*)d_in[4];
  const float* conv_b    = (const float*)d_in[5];
  const float* x_proj_w  = (const float*)d_in[6];
  const float* dt_w      = (const float*)d_in[7];
  const float* dt_b      = (const float*)d_in[8];
  const float* A_log     = (const float*)d_in[9];
  const float* D_skip    = (const float*)d_in[10];
  const float* out_w     = (const float*)d_in[11];
  const float* fnorm_w   = (const float*)d_in[12];

  char* ws = (char*)d_ws; size_t o = 0;
  auto take = [&](size_t bytes) -> char* { char* p = ws + o; o += bytes; return p; };

  float*  H     = (float*) take((size_t)ROWS * DMODEL * 4);        // residual stream
  bf16_t* XNBF  = (bf16_t*)take((size_t)ROWS * DMODEL * 2);        // rmsnorm(h) bf16
  float*  XZ    = (float*) take((size_t)ROWS * 2 * DINNER * 4);    // in_proj out
  float*  XC    = (float*) take((size_t)ROWS * DINNER * 4);        // conv+silu f32
  bf16_t* XCBF  = (bf16_t*)take((size_t)ROWS * DINNER * 2);        // conv+silu bf16
  float*  XDBL  = (float*) take((size_t)ROWS * XP_PAD * 4);        // x_proj out (stride 128)
  bf16_t* DTRBF = (bf16_t*)take((size_t)ROWS * DTRANK * 2);        // dtr bf16
  float*  DTRAW = (float*) take((size_t)ROWS * DINNER * 4);        // dt pre-softplus
  bf16_t* YBF   = (bf16_t*)take((size_t)ROWS * DINNER * 2);        // gated scan out
  bf16_t* WBF   = (bf16_t*)take((size_t)2 * DINNER * DMODEL * 2);  // weight staging
  bf16_t* EMBBF = (bf16_t*)take((size_t)VOCAB * DMODEL * 2);       // bf16 embed

  // h = embed[tokens]; convert embedding matrix once (reused as logits weight)
  embed_gather_kernel<<<ROWS, 256, 0, stream>>>(tokens, embed, H);
  launch_cvt(embed, EMBBF, (long)VOCAB * DMODEL, stream);

  for (int i = 0; i < NLAYER; ++i) {
    // xn = rmsnorm(h) -> bf16
    rmsnorm_bf16_kernel<<<ROWS, 256, 0, stream>>>(H, norm_w + (size_t)i * DMODEL, XNBF, DMODEL);
    // xz = xn @ in_w^T
    launch_cvt(in_proj_w + (size_t)i * 2 * DINNER * DMODEL, WBF, (long)2 * DINNER * DMODEL, stream);
    launch_gemm(XNBF, WBF, XZ, ROWS, 2 * DINNER, DMODEL, false, stream);
    // xc = silu(causal depthwise conv(xi) + cb)
    conv_silu_kernel<<<(ROWS * DINNER + 255) / 256, 256, 0, stream>>>(
        XZ, conv_w + (size_t)i * DINNER * DCONV, conv_b + (size_t)i * DINNER, XC, XCBF);
    // x_dbl = xc @ xp_w^T, weight zero-padded 96 -> 128 rows so N%64==0
    {
      long ntot = (long)XP_PAD * DINNER;
      f32_to_bf16_pad_kernel<<<(unsigned)((ntot + 255) / 256), 256, 0, stream>>>(
          x_proj_w + (size_t)i * 96 * DINNER, WBF, 96, DINNER, ntot);
      launch_gemm(XCBF, WBF, XDBL, ROWS, XP_PAD, DINNER, false, stream);
    }
    // dt_raw = dtr @ dt_w^T   (softplus + bias fused into the scan)
    f32_to_bf16_strided_kernel<<<(ROWS * DTRANK + 255) / 256, 256, 0, stream>>>(
        XDBL, DTRBF, ROWS, DTRANK, XP_PAD);
    launch_cvt(dt_w + (size_t)i * DINNER * DTRANK, WBF, (long)DINNER * DTRANK, stream);
    launch_gemm(DTRBF, WBF, DTRAW, ROWS, DINNER, DTRANK, false, stream);
    // selective scan + D-skip + silu(z) gating -> bf16
    ssm_scan_kernel<<<dim3(DINNER / 256, BATCH), 256, 0, stream>>>(
        XC, XDBL, DTRAW, dt_b + (size_t)i * DINNER, A_log + (size_t)i * DINNER * DSTATE,
        D_skip + (size_t)i * DINNER, XZ, YBF);
    // h += y @ ow^T   (fused residual via ADD epilogue)
    launch_cvt(out_w + (size_t)i * DMODEL * DINNER, WBF, (long)DMODEL * DINNER, stream);
    launch_gemm(YBF, WBF, H, ROWS, DMODEL, DINNER, true, stream);
  }

  // logits = rmsnorm(h, final_norm) @ embed^T
  rmsnorm_bf16_kernel<<<ROWS, 256, 0, stream>>>(H, fnorm_w, XNBF, DMODEL);
  launch_gemm(XNBF, EMBBF, (float*)d_out, ROWS, VOCAB, DMODEL, false, stream);
}
